// DaConA_3805341024787
// MI455X (gfx1250) — matrix operand order repaired and checked by reading, not compiled, not run on hardware
//
#include <hip/hip_runtime.h>
#include <hip/hip_bf16.h>
#include <math.h>

typedef __bf16 bf16;
typedef __attribute__((ext_vector_type(16))) bf16  v16bf;
typedef __attribute__((ext_vector_type(8)))  bf16  v8bf;
typedef __attribute__((ext_vector_type(4)))  bf16  v4bf;
typedef __attribute__((ext_vector_type(8)))  float v8f;

#define DIM_C 960
#define DIM_F 1024
#define H1    512
#define H2    256
#define H3    128

// ---------------- WMMA fragment helpers (contiguous k order, same for A and B) ----

static __device__ __forceinline__ v16bf join8(v8bf lo, v8bf hi) {
  v16bf r;
#pragma unroll
  for (int i = 0; i < 8; ++i) { r[i] = lo[i]; r[i + 8] = hi[i]; }
  return r;
}

// A (16xK tile, row-major source, ld elements per row):
// lane<16  : row=lane,    elems 0..15 = K kb+0..15
// lane>=16 : row=lane-16, elems 0..15 = K kb+16..31
static __device__ __forceinline__ v16bf load_a_frag(const bf16* base, int ld, int kb,
                                                    unsigned lane) {
  const unsigned m = lane & 15u, half = lane >> 4;
  const bf16* p = base + (size_t)m * ld + kb + half * 16;
  return join8(*(const v8bf*)p, *(const v8bf*)(p + 8));
}

// B (Kx16 tile) sourced from torch-style weight W[out][in] (ld = in-dim):
// lane<16 holds K=kb..kb+15 of column n=lane&15, lane>=16 holds K=kb+16..kb+31.
// One contiguous 32-byte load per lane.
static __device__ __forceinline__ v16bf load_b_frag(const bf16* w, int ld, int ob, int kb,
                                                    unsigned lane) {
  const unsigned n = lane & 15u, half = lane >> 4;
  return *(const v16bf*)(w + (size_t)(ob + n) * ld + kb + half * 16);
}

static __device__ __forceinline__ v8f wmma_bf16(v16bf a, v16bf b, v8f c) {
  return __builtin_amdgcn_wmma_f32_16x16x32_bf16(false, a, false, b, (short)0, c,
                                                 false, false);
}

// ---------------- kernel 0: fp32 -> bf16 weight conversion ---------------------

__global__ void k_cvt(const float* __restrict__ src, bf16* __restrict__ dst, int n) {
  int i = blockIdx.x * blockDim.x + threadIdx.x;
  const int stride = gridDim.x * blockDim.x;
  for (; i < n; i += stride) dst[i] = (bf16)src[i];
}

// ---------------- kernel 1: gather + dual transfer GEMM + elementwise product --
//
// Per wave: 16 batch rows. Both gathered operands (u_c, i_c) staged to LDS as
// bf16; tu and ti accumulated *together* so each B fragment of Wt is loaded
// once and feeds two independent WMMAs (better XDL pipelining, half the L2
// traffic of a two-pass design, no tu round-trip through LDS).

static __device__ __forceinline__ void stage16_rows(bf16* __restrict__ sdst,
                                                    const float* __restrict__ src,
                                                    const int* __restrict__ idx,
                                                    int r0, unsigned tid,
                                                    unsigned nthreads) {
  // 16 rows x DIM_C f32 -> bf16, vectorized by float4 (row stride 3840B)
  for (int i = tid; i < 16 * (DIM_C / 4); i += nthreads) {
    const int row = i / (DIM_C / 4);
    const int c4  = i % (DIM_C / 4);
    const float4 v = ((const float4*)(src + (size_t)idx[r0 + row] * DIM_C))[c4];
    v4bf o; o[0] = (bf16)v.x; o[1] = (bf16)v.y; o[2] = (bf16)v.z; o[3] = (bf16)v.w;
    *(v4bf*)(sdst + row * DIM_C + c4 * 4) = o;
  }
}

__global__ __launch_bounds__(64) void k_transfer(
    const int* __restrict__ rows, const int* __restrict__ cols,
    const float* __restrict__ user_inter, const float* __restrict__ item_inter,
    const float* __restrict__ user_indep, const float* __restrict__ item_indep,
    const bf16* __restrict__ Wtb, const float* __restrict__ bt,
    bf16* __restrict__ fac) {
  __shared__ __align__(32) bf16 sAu[2][16 * DIM_C];  // 60 KB
  __shared__ __align__(32) bf16 sAi[2][16 * DIM_C];  // 60 KB  (120 KB total)
  const unsigned w = threadIdx.x >> 5, lane = threadIdx.x & 31u;
  const int r0 = blockIdx.x * 32 + w * 16;
  const unsigned n = lane & 15u;
  const unsigned rowoff = (lane >> 4) * 8;

  // --- stage u_c and i_c for this wave's 16 rows ---
  stage16_rows(sAu[w], user_inter, rows, r0, lane, 32);
  stage16_rows(sAi[w], item_inter, cols, r0, lane, 32);
  __syncthreads();

  // --- fused: fac_inter = (u_c Wt^T + bt) * (i_c Wt^T + bt) ---
  for (int ob = 0; ob < DIM_C / 16; ++ob) {
    v8f au = {}, ai = {};
    for (int kb = 0; kb < DIM_C; kb += 32) {
      const v16bf b = load_b_frag(Wtb, DIM_C, ob * 16, kb, lane);
      au = wmma_bf16(load_a_frag(sAu[w], DIM_C, kb, lane), b, au);
      ai = wmma_bf16(load_a_frag(sAi[w], DIM_C, kb, lane), b, ai);
    }
    const float bias = bt[ob * 16 + n];
#pragma unroll
    for (int r = 0; r < 8; ++r) {
      const float v = (au[r] + bias) * (ai[r] + bias);
      fac[(size_t)(r0 + r + rowoff) * DIM_F + 64 + ob * 16 + n] = (bf16)v;
    }
  }

  // --- concatenated indep features at cols [0,64) for all 32 block rows ---
  const int rb = blockIdx.x * 32;
  for (int i = threadIdx.x; i < 32 * 32; i += 64) {
    const int row = i >> 5, k = i & 31;
    fac[(size_t)(rb + row) * DIM_F + k] =
        (bf16)user_indep[(size_t)rows[rb + row] * 32 + k];
    fac[(size_t)(rb + row) * DIM_F + 32 + k] =
        (bf16)item_indep[(size_t)cols[rb + row] * 32 + k];
  }
}

// ---------------- kernel 2: fused 3-layer tanh MLP + regression head -----------

__global__ __launch_bounds__(64) void k_mlp(
    const bf16* __restrict__ fac,
    const bf16* __restrict__ W1b, const float* __restrict__ b1,
    const bf16* __restrict__ W2b, const float* __restrict__ b2,
    const bf16* __restrict__ W3b, const float* __restrict__ b3,
    const float* __restrict__ Wr, const float* __restrict__ br,
    float* __restrict__ out) {
  __shared__ __align__(32) bf16 sF[2][16 * DIM_F];  // 64 KB
  __shared__ __align__(32) bf16 s1[2][16 * H1];     // 32 KB
  __shared__ __align__(32) bf16 s2[2][16 * H2];     // 16 KB
  __shared__ __align__(32) bf16 s3[2][16 * H3];     //  8 KB   (120 KB total)
  const unsigned w = threadIdx.x >> 5, lane = threadIdx.x & 31u;
  const int r0 = blockIdx.x * 32 + w * 16;
  const unsigned n = lane & 15u;
  const unsigned rowoff = (lane >> 4) * 8;

  // --- stage this wave's 16x1024 factor panel (contiguous 32 KB) into LDS ---
  {
    const v8bf* src = (const v8bf*)(fac + (size_t)r0 * DIM_F);
    v8bf* dst = (v8bf*)sF[w];
    for (int i = lane; i < 16 * DIM_F / 8; i += 32) dst[i] = src[i];
  }
  __syncthreads();

  // layer 1: [16,1024] x [1024,512] -> tanh -> s1
  for (int ob = 0; ob < H1 / 16; ++ob) {
    v8f acc = {};
    for (int kb = 0; kb < DIM_F; kb += 32)
      acc = wmma_bf16(load_a_frag(sF[w], DIM_F, kb, lane),
                      load_b_frag(W1b, DIM_F, ob * 16, kb, lane), acc);
    const float bias = b1[ob * 16 + n];
#pragma unroll
    for (int r = 0; r < 8; ++r)
      s1[w][(r + rowoff) * H1 + ob * 16 + n] = (bf16)tanhf(acc[r] + bias);
  }
  __syncthreads();

  // layer 2: [16,512] x [512,256] -> tanh -> s2
  for (int ob = 0; ob < H2 / 16; ++ob) {
    v8f acc = {};
    for (int kb = 0; kb < H1; kb += 32)
      acc = wmma_bf16(load_a_frag(s1[w], H1, kb, lane),
                      load_b_frag(W2b, H1, ob * 16, kb, lane), acc);
    const float bias = b2[ob * 16 + n];
#pragma unroll
    for (int r = 0; r < 8; ++r)
      s2[w][(r + rowoff) * H2 + ob * 16 + n] = (bf16)tanhf(acc[r] + bias);
  }
  __syncthreads();

  // layer 3: [16,256] x [256,128] -> tanh -> s3
  for (int ob = 0; ob < H3 / 16; ++ob) {
    v8f acc = {};
    for (int kb = 0; kb < H2; kb += 32)
      acc = wmma_bf16(load_a_frag(s2[w], H2, kb, lane),
                      load_b_frag(W3b, H2, ob * 16, kb, lane), acc);
    const float bias = b3[ob * 16 + n];
#pragma unroll
    for (int r = 0; r < 8; ++r)
      s3[w][(r + rowoff) * H3 + ob * 16 + n] = (bf16)tanhf(acc[r] + bias);
  }
  __syncthreads();

  // head: pred = a3 @ Wr^T + br + 3.5
  if (lane < 16) {
    float sum = br[0];
    const bf16* a = s3[w] + lane * H3;
#pragma unroll 8
    for (int k = 0; k < H3; ++k) sum += (float)a[k] * Wr[k];
    out[r0 + lane] = sum + 3.5f;
  }
}

// ---------------- launch -------------------------------------------------------

extern "C" void kernel_launch(void* const* d_in, const int* in_sizes, int n_in,
                              void* d_out, int out_size, void* d_ws, size_t ws_size,
                              hipStream_t stream) {
  const int*   rows       = (const int*)d_in[0];
  const int*   cols       = (const int*)d_in[1];
  const float* user_inter = (const float*)d_in[2];
  const float* item_inter = (const float*)d_in[3];
  const float* user_indep = (const float*)d_in[4];
  const float* item_indep = (const float*)d_in[5];
  const float* Wt         = (const float*)d_in[6];
  const float* bt         = (const float*)d_in[7];
  const float* W1         = (const float*)d_in[8];
  const float* b1         = (const float*)d_in[9];
  const float* W2         = (const float*)d_in[10];
  const float* b2         = (const float*)d_in[11];
  const float* W3         = (const float*)d_in[12];
  const float* b3         = (const float*)d_in[13];
  const float* Wr         = (const float*)d_in[14];
  const float* br         = (const float*)d_in[15];
  float* out = (float*)d_out;

  const int B = in_sizes[0];  // 131072

  // workspace layout (bytes)
  constexpr size_t WT_N = (size_t)DIM_C * DIM_C;  // 921600
  constexpr size_t W1_N = (size_t)H1 * DIM_F;     // 524288
  constexpr size_t W2_N = (size_t)H2 * H1;        // 131072
  constexpr size_t W3_N = (size_t)H3 * H2;        // 32768
  constexpr size_t WT_OFF  = 0;
  constexpr size_t W1_OFF  = WT_OFF + WT_N * 2;
  constexpr size_t W2_OFF  = W1_OFF + W1_N * 2;
  constexpr size_t W3_OFF  = W2_OFF + W2_N * 2;
  constexpr size_t FAC_OFF = W3_OFF + W3_N * 2;   // 3219456, 256B aligned

  char* ws   = (char*)d_ws;
  bf16* Wtb  = (bf16*)(ws + WT_OFF);
  bf16* W1b  = (bf16*)(ws + W1_OFF);
  bf16* W2b  = (bf16*)(ws + W2_OFF);
  bf16* W3b  = (bf16*)(ws + W3_OFF);
  bf16* fac  = (bf16*)(ws + FAC_OFF);

  k_cvt<<<256, 256, 0, stream>>>(Wt, Wtb, (int)WT_N);
  k_cvt<<<256, 256, 0, stream>>>(W1, W1b, (int)W1_N);
  k_cvt<<<128, 256, 0, stream>>>(W2, W2b, (int)W2_N);
  k_cvt<<<64,  256, 0, stream>>>(W3, W3b, (int)W3_N);

  k_transfer<<<B / 32, 64, 0, stream>>>(rows, cols, user_inter, item_inter,
                                        user_indep, item_indep, Wtb, bt, fac);

  k_mlp<<<B / 32, 64, 0, stream>>>(fac, W1b, b1, W2b, b2, W3b, b3, Wr, br, out);
}
